// Attention_6219112644956
// MI455X (gfx1250) — matrix operand order repaired and checked by reading, not compiled, not run on hardware
//
#include <hip/hip_runtime.h>

// ---------------------------------------------------------------------------
// Types for CDNA5 WMMA (wave32): A/B = 16 bf16 per lane (16x32 / 32x16 tile),
// C/D = 8 f32 per lane (16x16 tile).
// ---------------------------------------------------------------------------
typedef __bf16 bf16_t;
typedef bf16_t v16bf __attribute__((ext_vector_type(16)));
typedef float  v8f   __attribute__((ext_vector_type(8)));
typedef unsigned int u32x4 __attribute__((ext_vector_type(4)));

union FragU {
  v16bf v;
  u32x4 q[2];
  unsigned int d[8];
};

__device__ __forceinline__ unsigned int f2bf1(float f) {
  // round-to-nearest-even f32 -> bf16
  unsigned int u = __float_as_uint(f);
  return (u + 0x7FFFu + ((u >> 16) & 1u)) >> 16;
}
__device__ __forceinline__ unsigned int packbf(float lo, float hi) {
  return f2bf1(lo) | (f2bf1(hi) << 16);
}

// A-operand fragment: lane<16 -> row (lane&15), elems 0..7 = K off..off+7,
// elems 8..15 = K off+16..off+23 ; lane>=16 handled by caller via `off` (+8).
__device__ __forceinline__ v16bf frag_split(const unsigned short* p, int off) {
  FragU f;
  f.q[0] = *(const u32x4*)(p + off);
  f.q[1] = *(const u32x4*)(p + off + 16);
  return f.v;
}
// A-operand fragment: lane<16 -> row (lane&15) K=0..15 contiguous,
// lane>=16 -> same row K=16..31 ; caller passes off = 0 or 16 (+dc).
__device__ __forceinline__ v16bf frag_contig(const unsigned short* p, int off) {
  FragU f;
  f.q[0] = *(const u32x4*)(p + off);
  f.q[1] = *(const u32x4*)(p + off + 8);
  return f.v;
}

__device__ __forceinline__ v8f wmma_bf16(v16bf a, v16bf b, v8f c) {
  return __builtin_amdgcn_wmma_f32_16x16x32_bf16(
      /*neg_a=*/false, a, /*neg_b=*/false, b,
      /*c_mod=*/(short)0, c, /*reuse_a=*/false, /*reuse_b=*/false);
}

// gfx1250 async global->LDS copy (16B per lane), tracked by ASYNCcnt.
// VDST VGPR carries the LDS byte offset (low 32 bits of the generic pointer).
__device__ __forceinline__ void async_load_b128(unsigned int lds_addr,
                                                const void* gaddr) {
  asm volatile("global_load_async_to_lds_b128 %0, %1, off"
               :: "v"(lds_addr), "v"(gaddr)
               : "memory");
}
__device__ __forceinline__ unsigned int lds_off(const void* p) {
  return (unsigned int)(size_t)p;  // AS3->generic cast keeps offset in low 32b
}

// Problem constants: B=2, N=2048, C=768, H=12, D=64
#define SEQ_N 2048
#define CH    768
#define HEADS 12
#define HDIM  64
#define ROWS  4096           // B * N

// ---------------------------------------------------------------------------
// Kernel 1: qkv = x @ w_qkv  (f32 in, bf16 out, split into Q / K / V^T)
// Workgroup: 256 threads (8 waves), 128x64 output tile, K-step 32.
// ---------------------------------------------------------------------------
__global__ __launch_bounds__(256) void qkv_gemm(
    const float* __restrict__ x, const float* __restrict__ w,
    unsigned short* __restrict__ qw, unsigned short* __restrict__ kw,
    unsigned short* __restrict__ vtw) {
  __shared__ __align__(16) unsigned short ldsA[128 * 40];   // [m][k] pad 40
  __shared__ __align__(16) unsigned short ldsBt[64 * 40];   // [n][k] pad 40
  const int tid = threadIdx.x;
  const int lane = tid & 31;
  const int wave = tid >> 5;
  const int rb = blockIdx.y * 128;   // row block in [0,4096)
  const int cb = blockIdx.x * 64;    // col block in [0,2304)

  v8f acc[4];
#pragma unroll
  for (int t = 0; t < 4; ++t)
#pragma unroll
    for (int j = 0; j < 8; ++j) acc[t][j] = 0.0f;

  const int arow_s = tid >> 1, ac0 = (tid & 1) * 16;
  const int bcol_s = tid & 63, bk0 = (tid >> 6) * 8;

  for (int kb = 0; kb < CH; kb += 32) {
    {  // stage A tile 128x32 (f32 -> bf16)
      const float* src = x + (size_t)(rb + arow_s) * CH + kb + ac0;
      unsigned int pk[8];
#pragma unroll
      for (int j = 0; j < 8; ++j) pk[j] = packbf(src[2 * j], src[2 * j + 1]);
      u32x4 lo = {pk[0], pk[1], pk[2], pk[3]};
      u32x4 hi = {pk[4], pk[5], pk[6], pk[7]};
      *(u32x4*)&ldsA[arow_s * 40 + ac0] = lo;
      *(u32x4*)&ldsA[arow_s * 40 + ac0 + 8] = hi;
    }
    {  // stage B tile 32x64 transposed -> ldsBt[col][k]
      const float* src = w + (size_t)(kb + bk0) * 2304 + cb + bcol_s;
      unsigned int pk[4];
#pragma unroll
      for (int u = 0; u < 4; ++u)
        pk[u] = packbf(src[(size_t)(2 * u) * 2304], src[(size_t)(2 * u + 1) * 2304]);
      u32x4 v = {pk[0], pk[1], pk[2], pk[3]};
      *(u32x4*)&ldsBt[bcol_s * 40 + bk0] = v;
    }
    __syncthreads();

    const int am = wave * 16 + (lane & 15);
    const int aoff = (lane < 16) ? 0 : 8;
    v16bf afrag = frag_split(&ldsA[am * 40], aoff);
    const int bcol = lane & 15;
    const int bko = (lane < 16) ? 0 : 8;
#pragma unroll
    for (int nt = 0; nt < 4; ++nt) {
      v16bf bfrag = frag_split(&ldsBt[(nt * 16 + bcol) * 40], bko);
      acc[nt] = wmma_bf16(afrag, bfrag, acc[nt]);
    }
    __syncthreads();
  }

  // Epilogue: 64-wide col tile lies entirely within one (s, h) group.
  const int s = cb / CH;                 // 0=Q, 1=K, 2=V
  const int h = (cb % CH) / HDIM;
  const int b = rb / SEQ_N;
  const int nloc = (rb % SEQ_N) + wave * 16 + ((lane < 16) ? 0 : 8);
  const size_t bh = (size_t)(b * HEADS + h);
  if (s == 2) {  // V stored transposed: vt[bh][d][n] -> contiguous 16B stores
#pragma unroll
    for (int nt = 0; nt < 4; ++nt) {
      const int d = nt * 16 + (lane & 15);
      unsigned short* dst = vtw + ((bh * HDIM + d) * SEQ_N + nloc);
      unsigned int pk[4];
#pragma unroll
      for (int r = 0; r < 4; ++r) pk[r] = packbf(acc[nt][2 * r], acc[nt][2 * r + 1]);
      u32x4 v = {pk[0], pk[1], pk[2], pk[3]};
      *(u32x4*)dst = v;
    }
  } else {       // Q (scaled by D^-0.5) / K stored [bh][n][d]
    unsigned short* base = (s == 0) ? qw : kw;
    const float mult = (s == 0) ? 0.125f : 1.0f;  // 1/sqrt(64)
#pragma unroll
    for (int nt = 0; nt < 4; ++nt) {
      const int d = nt * 16 + (lane & 15);
#pragma unroll
      for (int r = 0; r < 8; ++r)
        base[(bh * SEQ_N + (size_t)(nloc + r)) * HDIM + d] =
            (unsigned short)f2bf1(acc[nt][r] * mult);
    }
  }
}

// ---------------------------------------------------------------------------
// Kernel 2: flash attention with async double-buffered K/V^T staging.
// Each wave owns 16 queries, streams 32-key chunks. S^T = K·Q^T (keys in
// VGPRs, queries in lanes) so the online softmax is per-lane + one lane^16
// exchange, and the S^T accumulator maps onto the B-operand of O^T = V^T·P^T
// with one half-fragment lane^16 swap. K/V^T tiles for chunk i+1 stream into
// LDS via global_load_async_to_lds_b128 while chunk i computes.
// ---------------------------------------------------------------------------
#define KBUF (32 * 72)
#define VBUF (64 * 40)

__global__ __launch_bounds__(256) void flash_attn(
    const unsigned short* __restrict__ qw, const unsigned short* __restrict__ kw,
    const unsigned short* __restrict__ vtw, unsigned short* __restrict__ ow) {
  __shared__ __align__(16) unsigned short ldsK[2 * KBUF];    // [key][d] pad 72
  __shared__ __align__(16) unsigned short ldsVT[2 * VBUF];   // [d][key] pad 40
  const int tid = threadIdx.x;
  const int lane = tid & 31;
  const int wave = tid >> 5;
  const int bh = blockIdx.y;                 // [0,24)
  const int b = bh / HEADS, h = bh % HEADS;
  const int q0 = blockIdx.x * 128 + wave * 16;

  const unsigned short* qp = qw + (size_t)bh * SEQ_N * HDIM;
  const unsigned short* kp = kw + (size_t)bh * SEQ_N * HDIM;
  const unsigned short* vp = vtw + (size_t)bh * HDIM * SEQ_N;

  // Q^T B-fragments (held in registers for the whole kv loop)
  const int qrow = q0 + (lane & 15);
  const int qo = (lane < 16) ? 0 : 8;
  v16bf qf0 = frag_split(qp + (size_t)qrow * HDIM, 0 + qo);
  v16bf qf1 = frag_split(qp + (size_t)qrow * HDIM, 32 + qo);

  v8f oacc[4];
#pragma unroll
  for (int t = 0; t < 4; ++t)
#pragma unroll
    for (int j = 0; j < 8; ++j) oacc[t][j] = 0.0f;
  float mi = -1e30f, li = 0.0f;

  const int skr = tid >> 3, skc = (tid & 7) * 8;  // K staging: 32x64 bf16
  const int svd = tid >> 2, svc = (tid & 3) * 8;  // VT staging: 64x32 bf16
  const int koff = (lane < 16) ? 0 : 8;
  const int krow = lane & 15;

  const unsigned int ldsK_base = lds_off(&ldsK[skr * 72 + skc]);
  const unsigned int ldsVT_base = lds_off(&ldsVT[svd * 40 + svc]);

  auto issue_chunk = [&](int kb, int sel) {
    async_load_b128(ldsK_base + (unsigned)sel * (KBUF * 2),
                    kp + (size_t)(kb + skr) * HDIM + skc);
    async_load_b128(ldsVT_base + (unsigned)sel * (VBUF * 2),
                    vp + (size_t)svd * SEQ_N + kb + svc);
  };

  issue_chunk(0, 0);
  int sel = 0;

  for (int kb = 0; kb < SEQ_N; kb += 32) {
    if (kb + 32 < SEQ_N) {
      issue_chunk(kb + 32, sel ^ 1);
      // 2 just-issued ops may stay outstanding => previous buffer complete.
      asm volatile("s_wait_asynccnt 0x2" ::: "memory");
    } else {
      asm volatile("s_wait_asynccnt 0x0" ::: "memory");
    }
    __syncthreads();

    const unsigned short* ldsKs = &ldsK[sel * KBUF];
    const unsigned short* ldsVs = &ldsVT[sel * VBUF];

    // S^T tiles: s0 = keys kb..kb+15, s1 = keys kb+16..kb+31 (x 16 queries)
    v8f s0, s1;
#pragma unroll
    for (int j = 0; j < 8; ++j) { s0[j] = 0.0f; s1[j] = 0.0f; }
    s0 = wmma_bf16(frag_split(&ldsKs[krow * 72], koff), qf0, s0);
    s0 = wmma_bf16(frag_split(&ldsKs[krow * 72], 32 + koff), qf1, s0);
    s1 = wmma_bf16(frag_split(&ldsKs[(16 + krow) * 72], koff), qf0, s1);
    s1 = wmma_bf16(frag_split(&ldsKs[(16 + krow) * 72], 32 + koff), qf1, s1);

    // online softmax: each lane holds 16 scores of one query; partner lane^16
    // holds the other 16 of the same 32-key chunk.
    float cmax = s0[0];
#pragma unroll
    for (int j = 1; j < 8; ++j) cmax = fmaxf(cmax, s0[j]);
#pragma unroll
    for (int j = 0; j < 8; ++j) cmax = fmaxf(cmax, s1[j]);
    cmax = fmaxf(cmax, __shfl_xor(cmax, 16, 32));
    const float mnew = fmaxf(mi, cmax);
    const float alpha = __expf(mi - mnew);
    mi = mnew;

    float p0[8], p1[8], ps = 0.0f;
#pragma unroll
    for (int j = 0; j < 8; ++j) { p0[j] = __expf(s0[j] - mnew); ps += p0[j]; }
#pragma unroll
    for (int j = 0; j < 8; ++j) { p1[j] = __expf(s1[j] - mnew); ps += p1[j]; }
    ps += __shfl_xor(ps, 16, 32);
    li = li * alpha + ps;

#pragma unroll
    for (int t = 0; t < 4; ++t)
#pragma unroll
      for (int j = 0; j < 8; ++j) oacc[t][j] *= alpha;

    // Build P^T B-fragment: lane<16 needs keys 0..15 (own p0 + partner p0),
    // lane>=16 needs keys 16..31 (partner p1 + own p1).
    unsigned int pk0[4], pk1[4];
#pragma unroll
    for (int r = 0; r < 4; ++r) {
      pk0[r] = packbf(p0[2 * r], p0[2 * r + 1]);
      pk1[r] = packbf(p1[2 * r], p1[2 * r + 1]);
    }
    FragU pf;
#pragma unroll
    for (int r = 0; r < 4; ++r) {
      const unsigned int o0 = __shfl_xor(pk0[r], 16, 32);
      const unsigned int o1 = __shfl_xor(pk1[r], 16, 32);
      pf.d[r] = (lane < 16) ? pk0[r] : o1;
      pf.d[4 + r] = (lane < 16) ? o0 : pk1[r];
    }

    // O^T += V^T(tile) x P^T : 4 d-tiles of 16
#pragma unroll
    for (int dt = 0; dt < 4; ++dt) {
      v16bf va = frag_contig(&ldsVs[(dt * 16 + (lane & 15)) * 40], (lane < 16) ? 0 : 16);
      oacc[dt] = wmma_bf16(va, pf.v, oacc[dt]);
    }
    sel ^= 1;
    __syncthreads();  // all ds reads of this buffer retired before reuse
  }

  // Normalize and store O into [ROWS, CH] bf16 activation buffer.
  const float inv = 1.0f / li;
  const int nrow = b * SEQ_N + q0 + (lane & 15);
#pragma unroll
  for (int dt = 0; dt < 4; ++dt) {
    const int d0 = dt * 16 + ((lane < 16) ? 0 : 8);
    unsigned short* dst = ow + ((size_t)nrow * CH + h * HDIM + d0);
    unsigned int pk[4];
#pragma unroll
    for (int r = 0; r < 4; ++r)
      pk[r] = packbf(oacc[dt][2 * r] * inv, oacc[dt][2 * r + 1] * inv);
    u32x4 v = {pk[0], pk[1], pk[2], pk[3]};
    *(u32x4*)dst = v;
  }
}

// ---------------------------------------------------------------------------
// Kernel 3: out = attn_out(bf16) @ w_proj + b_proj  (f32 output)
// A tile is a pure bf16 copy -> use async global->LDS path.
// ---------------------------------------------------------------------------
__global__ __launch_bounds__(256) void proj_gemm(
    const unsigned short* __restrict__ a, const float* __restrict__ w,
    const float* __restrict__ bias, float* __restrict__ out) {
  __shared__ __align__(16) unsigned short ldsA[128 * 40];
  __shared__ __align__(16) unsigned short ldsBt[64 * 40];
  const int tid = threadIdx.x;
  const int lane = tid & 31;
  const int wave = tid >> 5;
  const int rb = blockIdx.y * 128, cb = blockIdx.x * 64;

  v8f acc[4];
#pragma unroll
  for (int t = 0; t < 4; ++t)
#pragma unroll
    for (int j = 0; j < 8; ++j) acc[t][j] = 0.0f;

  const int arow_s = tid >> 1, ac0 = (tid & 1) * 16;
  const int bcol_s = tid & 63, bk0 = (tid >> 6) * 8;
  const unsigned int ldsA_addr = lds_off(&ldsA[arow_s * 40 + ac0]);

  for (int kb = 0; kb < CH; kb += 32) {
    {  // stage A (already bf16): two async 16B global->LDS copies per thread
      const unsigned short* src = a + (size_t)(rb + arow_s) * CH + kb + ac0;
      async_load_b128(ldsA_addr, src);
      async_load_b128(ldsA_addr + 16, src + 8);
    }
    {  // stage B transposed (f32 -> bf16)
      const float* src = w + (size_t)(kb + bk0) * CH + cb + bcol_s;
      unsigned int pk[4];
#pragma unroll
      for (int u = 0; u < 4; ++u)
        pk[u] = packbf(src[(size_t)(2 * u) * CH], src[(size_t)(2 * u + 1) * CH]);
      u32x4 v = {pk[0], pk[1], pk[2], pk[3]};
      *(u32x4*)&ldsBt[bcol_s * 40 + bk0] = v;
    }
    asm volatile("s_wait_asynccnt 0x0" ::: "memory");
    __syncthreads();

    const int am = wave * 16 + (lane & 15);
    const int aoff = (lane < 16) ? 0 : 8;
    v16bf afrag = frag_split(&ldsA[am * 40], aoff);
    const int bcol = lane & 15;
    const int bko = (lane < 16) ? 0 : 8;
#pragma unroll
    for (int nt = 0; nt < 4; ++nt) {
      v16bf bfrag = frag_split(&ldsBt[(nt * 16 + bcol) * 40], bko);
      acc[nt] = wmma_bf16(afrag, bfrag, acc[nt]);
    }
    __syncthreads();
  }

  const int mrow = rb + wave * 16 + ((lane < 16) ? 0 : 8);
#pragma unroll
  for (int nt = 0; nt < 4; ++nt) {
    const int c = cb + nt * 16 + (lane & 15);
    const float bv = bias[c];
#pragma unroll
    for (int r = 0; r < 8; ++r)
      out[(size_t)(mrow + r) * CH + c] = acc[nt][r] + bv;
  }
}

// ---------------------------------------------------------------------------
extern "C" void kernel_launch(void* const* d_in, const int* in_sizes, int n_in,
                              void* d_out, int out_size, void* d_ws, size_t ws_size,
                              hipStream_t stream) {
  const float* x = (const float*)d_in[0];       // [2,2048,768]
  const float* w_qkv = (const float*)d_in[1];   // [768,2304]
  const float* w_proj = (const float*)d_in[2];  // [768,768]
  const float* b_proj = (const float*)d_in[3];  // [768]
  float* out = (float*)d_out;                   // [2,2048,768] f32

  const size_t per_buf = (size_t)2 * HEADS * SEQ_N * HDIM;  // 3,145,728 elems
  unsigned short* q_ws = (unsigned short*)d_ws;
  unsigned short* k_ws = q_ws + per_buf;
  unsigned short* vt_ws = k_ws + per_buf;
  unsigned short* o_ws = vt_ws + per_buf;       // [4096,768] bf16

  qkv_gemm<<<dim3(2304 / 64, ROWS / 128), 256, 0, stream>>>(x, w_qkv, q_ws, k_ws, vt_ws);
  flash_attn<<<dim3(SEQ_N / 128, 2 * HEADS), 256, 0, stream>>>(q_ws, k_ws, vt_ws, o_ws);
  proj_gemm<<<dim3(CH / 64, ROWS / 128), 256, 0, stream>>>(o_ws, w_proj, b_proj, out);
}
